// SelfAttentionRelPos_91336774517020
// MI455X (gfx1250) — matrix operand order repaired and checked by reading, not compiled, not run on hardware
//
#include <hip/hip_runtime.h>
#include <hip/hip_bf16.h>
#include <stdint.h>

// ---------------- CDNA5 WMMA types ----------------
typedef __attribute__((ext_vector_type(16))) __bf16 v16bf;
typedef __attribute__((ext_vector_type(8)))  __bf16 v8bf;
typedef __attribute__((ext_vector_type(8)))  float  v8f;

#define BATCH  8
#define NHEADS 12
#define NTOK   1025
#define NPAD   1056      // 66 tiles of 16, multiple of 32 for K-chunks
#define NQT    66
#define FTOK   768
#define FHEAD  64

__device__ __forceinline__ __bf16 f2bf(float x) {
  unsigned u = __builtin_bit_cast(unsigned, x);
  unsigned r = (u + 0x7fffu + ((u >> 16) & 1u)) >> 16;   // RNE
  return __builtin_bit_cast(__bf16, (unsigned short)r);
}

__device__ __forceinline__ v16bf combine8(v8bf lo, v8bf hi) {
  v16bf r;
#pragma unroll
  for (int i = 0; i < 8; ++i) { r[i] = lo[i]; r[i + 8] = hi[i]; }
  return r;
}

// A fragment (16x32 bf16, MxK). src -> element [0][0] of tile, row-major, `stride` elems.
// ISA 7.12.2: lane m = lane&15; lanes<16 hold K = j (j<8) / 16+j-8 ; lanes>=16 offset +8.
__device__ __forceinline__ v16bf load_frag_a(const __bf16* src, int stride, int lane) {
  int m   = lane & 15;
  int hi8 = (lane >> 4) << 3;                 // 0 or 8
  const __bf16* p = src + (long)m * stride;
  v8bf lo = *(const v8bf*)(p + hi8);
  v8bf hi = *(const v8bf*)(p + 16 + hi8);
  return combine8(lo, hi);
}

// B fragment (32x16 bf16, KxN) sourced from Bt = [n][k] row-major ("weights as rows").
// lane n = lane&15; lanes<16 hold K = j (j<8) / 16+j-8 ; lanes>=16 offset +8 (matches A).
__device__ __forceinline__ v16bf load_frag_b(const __bf16* srcT, int stride, int lane) {
  int n   = lane & 15;
  int k16 = (lane >> 4) << 3;                 // 0 or 8
  const __bf16* p = srcT + (long)n * stride + k16;
  v8bf lo = *(const v8bf*)(p);
  v8bf hi = *(const v8bf*)(p + 16);
  return combine8(lo, hi);
}

__device__ __forceinline__ v8f wmma_bf16(v16bf a, v16bf b, v8f c) {
  return __builtin_amdgcn_wmma_f32_16x16x32_bf16(false, a, false, b, (short)0, c,
                                                 false, false);
}

#define LDS_FENCE() asm volatile("s_wait_dscnt 0x0" ::: "memory")

// ---------------- conversion kernels ----------------
__global__ void k_cvt_tokens(const float* __restrict__ t, __bf16* __restrict__ o) {
  long i = (long)blockIdx.x * blockDim.x + threadIdx.x;
  int  c = (int)(i % FTOK);
  long nb = i / FTOK;
  int  n = (int)(nb % NPAD);
  int  b = (int)(nb / NPAD);
  float v = (n < NTOK) ? t[((long)b * NTOK + n) * FTOK + c] : 0.0f;
  o[i] = f2bf(v);
}

__global__ void k_cvt(const float* __restrict__ s, __bf16* __restrict__ d, int n) {
  int i = blockIdx.x * blockDim.x + threadIdx.x;
  if (i < n) d[i] = f2bf(s[i]);
}

// bias[h][nq][nk] = relpos_table[relpos_index[nq][nk]][h], zero in padding.
__global__ void k_bias(const int* __restrict__ rpi, const float* __restrict__ table,
                       float* __restrict__ bias) {
  long i = (long)blockIdx.x * blockDim.x + threadIdx.x;
  int  nk = (int)(i % NPAD);
  long t2 = i / NPAD;
  int  nq = (int)(t2 % NPAD);
  int  h  = (int)(t2 / NPAD);
  float v = 0.0f;
  if (nq < NTOK && nk < NTOK) v = table[(long)rpi[(long)nq * NTOK + nk] * NHEADS + h];
  bias[i] = v;
}

// ---------------- QKV projection: qkv = tokens @ qkv_w^T ----------------
// grid: B * NQT * 9 blocks of 128 (4 waves); wave -> one 16(row) x 64(col) tile.
__global__ void k_qkv(const __bf16* __restrict__ tok, const __bf16* __restrict__ w,
                      const float* __restrict__ qb, const float* __restrict__ vb,
                      __bf16* __restrict__ Q, __bf16* __restrict__ K,
                      __bf16* __restrict__ Vt) {
  int lane = threadIdx.x & 31, wave = threadIdx.x >> 5;
  int og = blockIdx.x % 9;
  int qt = (blockIdx.x / 9) % NQT;
  int b  = blockIdx.x / (9 * NQT);
  int o0 = (og * 4 + wave) * 64;
  const __bf16* arow = tok + ((long)b * NPAD + qt * 16) * FTOK;
  v8f acc[4] = {};
  for (int kc = 0; kc < FTOK; kc += 32) {
    v16bf af = load_frag_a(arow + kc, FTOK, lane);
#pragma unroll
    for (int ft = 0; ft < 4; ++ft) {
      v16bf bf = load_frag_b(w + (long)(o0 + ft * 16) * FTOK + kc, FTOK, lane);
      acc[ft] = wmma_bf16(af, bf, acc[ft]);
    }
  }
  int hi8 = (lane >> 4) << 3, col = lane & 15, m0 = qt * 16;
  if (o0 < FTOK) {                                   // Q: bias + 1/sqrt(64)
    int h = o0 / 64;
#pragma unroll
    for (int ft = 0; ft < 4; ++ft) {
      int f = ft * 16 + col;
      float bq = qb[h * 64 + f];
#pragma unroll
      for (int r = 0; r < 8; ++r) {
        int m = m0 + r + hi8;
        Q[(((long)b * NHEADS + h) * NPAD + m) * FHEAD + f] =
            f2bf((acc[ft][r] + bq) * 0.125f);
      }
    }
  } else if (o0 < 2 * FTOK) {                        // K: no bias
    int h = (o0 - FTOK) / 64;
#pragma unroll
    for (int ft = 0; ft < 4; ++ft) {
      int f = ft * 16 + col;
#pragma unroll
      for (int r = 0; r < 8; ++r) {
        int m = m0 + r + hi8;
        K[(((long)b * NHEADS + h) * NPAD + m) * FHEAD + f] = f2bf(acc[ft][r]);
      }
    }
  } else {                                           // V: bias, stored transposed [f][tok]
    int h = (o0 - 2 * FTOK) / 64;
#pragma unroll
    for (int ft = 0; ft < 4; ++ft) {
      int f = ft * 16 + col;
      float bv = vb[h * 64 + f];
#pragma unroll
      for (int r = 0; r < 8; ++r) {
        int m = m0 + r + hi8;
        Vt[(((long)b * NHEADS + h) * FHEAD + f) * NPAD + m] = f2bf(acc[ft][r] + bv);
      }
    }
  }
}

// ---------------- flash-style attention (two-pass, no S in HBM) ----------------
// grid: B*NHEADS*17 blocks of 128; each wave owns one 16-query tile + private LDS slice.
__global__ void k_attn(const __bf16* __restrict__ Q, const __bf16* __restrict__ K,
                       const __bf16* __restrict__ Vt, const float* __restrict__ bias,
                       __bf16* __restrict__ ctx) {
  __shared__ __attribute__((aligned(16))) float          Stile[4][256];
  __shared__ __attribute__((aligned(16))) unsigned short Pch[4][512];   // 16x32 bf16
  __shared__ float rowM[4][16];
  __shared__ float rowS[4][16];

  int lane = threadIdx.x & 31, wv = threadIdx.x >> 5;
  int bh = blockIdx.x / 17, qg = blockIdx.x % 17;
  int qt = qg * 4 + wv;
  if (qt >= NQT) return;                      // wave-uniform
  int b = bh / NHEADS, h = bh % NHEADS;
  long qkbase = ((long)b * NHEADS + h) * NPAD;
  int hi8 = (lane >> 4) << 3, col = lane & 15;

  const __bf16* qrow = Q + (qkbase + qt * 16) * FHEAD;
  v16bf qf0 = load_frag_a(qrow + 0,  FHEAD, lane);
  v16bf qf1 = load_frag_a(qrow + 32, FHEAD, lane);

  if (lane < 16) { rowM[wv][lane] = -3.0e37f; rowS[wv][lane] = 0.0f; }
  LDS_FENCE();

  const float* brow = bias + ((long)h * NPAD + qt * 16) * NPAD;

  // ---- pass 1: row max / sum-of-exp (online) ----
  for (int t = 0; t < 65; ++t) {              // tiles with any valid kv (<=1024+16)
    int kv0 = t * 16;
    v8f s = {};
    s = wmma_bf16(qf0, load_frag_b(K + (qkbase + kv0) * FHEAD + 0,  FHEAD, lane), s);
    s = wmma_bf16(qf1, load_frag_b(K + (qkbase + kv0) * FHEAD + 32, FHEAD, lane), s);
#pragma unroll
    for (int r = 0; r < 8; ++r)
      Stile[wv][(r + hi8) * 16 + col] = s[r] + brow[(long)(r + hi8) * NPAD + kv0 + col];
    LDS_FENCE();
    if (lane < 16) {
      int row = lane;
      float M = rowM[wv][row], S = rowS[wv][row];
      int nv = NTOK - kv0; if (nv > 16) nv = 16;
      float tmax = -3.0e37f;
      for (int c = 0; c < nv; ++c) tmax = fmaxf(tmax, Stile[wv][row * 16 + c]);
      float nM = fmaxf(M, tmax);
      float a = S * __expf(M - nM);
      for (int c = 0; c < nv; ++c) a += __expf(Stile[wv][row * 16 + c] - nM);
      rowM[wv][row] = nM; rowS[wv][row] = a;
    }
    LDS_FENCE();
  }

  // ---- pass 2: recompute S, P = exp(S-M) (bf16 in LDS), accumulate P @ V^T ----
  v8f o[4] = {};
  long vbase = ((long)b * NHEADS + h) * FHEAD;
  for (int ch = 0; ch < 33; ++ch) {           // K-chunks of 32 kv positions
#pragma unroll
    for (int sub = 0; sub < 2; ++sub) {
      int kv0 = (ch * 2 + sub) * 16;
      if (kv0 < NTOK) {
        v8f s = {};
        s = wmma_bf16(qf0, load_frag_b(K + (qkbase + kv0) * FHEAD + 0,  FHEAD, lane), s);
        s = wmma_bf16(qf1, load_frag_b(K + (qkbase + kv0) * FHEAD + 32, FHEAD, lane), s);
#pragma unroll
        for (int r = 0; r < 8; ++r)
          Stile[wv][(r + hi8) * 16 + col] =
              s[r] + brow[(long)(r + hi8) * NPAD + kv0 + col];
        LDS_FENCE();
        if (lane < 16) {
          int row = lane;
          float M = rowM[wv][row];
          for (int c = 0; c < 16; ++c) {
            float p = (kv0 + c < NTOK) ? __expf(Stile[wv][row * 16 + c] - M) : 0.0f;
            Pch[wv][row * 32 + sub * 16 + c] = __builtin_bit_cast(unsigned short, f2bf(p));
          }
        }
      } else if (lane < 16) {
        for (int c = 0; c < 16; ++c) Pch[wv][lane * 32 + sub * 16 + c] = 0;
      }
    }
    LDS_FENCE();
    v16bf pf = load_frag_a((const __bf16*)&Pch[wv][0], 32, lane);
#pragma unroll
    for (int ft = 0; ft < 4; ++ft) {
      v16bf vf = load_frag_b((const __bf16*)(Vt + (vbase + ft * 16) * NPAD + ch * 32),
                             NPAD, lane);
      o[ft] = wmma_bf16(pf, vf, o[ft]);
    }
  }

  // ---- normalize by row sums, emit bf16 context [b][tok][h*64+f] ----
#pragma unroll
  for (int ft = 0; ft < 4; ++ft) {
    int f = ft * 16 + col;
#pragma unroll
    for (int r = 0; r < 8; ++r) {
      int mg = qt * 16 + r + hi8;
      if (mg < NTOK) {
        float inv = 1.0f / rowS[wv][r + hi8];
        ctx[((long)b * NPAD + mg) * FTOK + h * FHEAD + f] = f2bf(o[ft][r] * inv);
      }
    }
  }
}

// ---------------- output projection: out = ctx @ proj_w^T + proj_b ----------------
__global__ void k_proj(const __bf16* __restrict__ ctx, const __bf16* __restrict__ w,
                       const float* __restrict__ pb, float* __restrict__ out) {
  int lane = threadIdx.x & 31, wv = threadIdx.x >> 5;
  int og = blockIdx.x % 3;
  int qt = (blockIdx.x / 3) % NQT;
  int b  = blockIdx.x / (3 * NQT);
  int o0 = (og * 4 + wv) * 64;
  const __bf16* arow = ctx + ((long)b * NPAD + qt * 16) * FTOK;
  v8f acc[4] = {};
  for (int kc = 0; kc < FTOK; kc += 32) {
    v16bf af = load_frag_a(arow + kc, FTOK, lane);
#pragma unroll
    for (int ft = 0; ft < 4; ++ft) {
      v16bf bf = load_frag_b(w + (long)(o0 + ft * 16) * FTOK + kc, FTOK, lane);
      acc[ft] = wmma_bf16(af, bf, acc[ft]);
    }
  }
  int hi8 = (lane >> 4) << 3, col = lane & 15;
#pragma unroll
  for (int ft = 0; ft < 4; ++ft) {
    int oc = o0 + ft * 16 + col;
    float bb = pb[oc];
#pragma unroll
    for (int r = 0; r < 8; ++r) {
      int mg = qt * 16 + r + hi8;
      if (mg < NTOK) out[((long)b * NTOK + mg) * FTOK + oc] = acc[ft][r] + bb;
    }
  }
}

// ---------------- launch ----------------
extern "C" void kernel_launch(void* const* d_in, const int* in_sizes, int n_in,
                              void* d_out, int out_size, void* d_ws, size_t ws_size,
                              hipStream_t stream) {
  const float* tokens  = (const float*)d_in[0];
  const float* qkv_w   = (const float*)d_in[1];
  const float* q_bias  = (const float*)d_in[2];
  const float* v_bias  = (const float*)d_in[3];
  const float* rtable  = (const float*)d_in[4];
  const float* proj_w  = (const float*)d_in[5];
  const float* proj_b  = (const float*)d_in[6];
  const int*   rpi     = (const int*)d_in[7];
  float* out = (float*)d_out;

  // workspace carve-up (all sizes 256B-multiples)
  char* ws = (char*)d_ws;
  const size_t SZ_TOK  = (size_t)BATCH * NPAD * FTOK * 2;          // 12.98 MB
  const size_t SZ_QKVW = (size_t)3 * FTOK * FTOK * 2;              // 3.54 MB
  const size_t SZ_PRJW = (size_t)FTOK * FTOK * 2;                  // 1.18 MB
  const size_t SZ_QKV  = (size_t)BATCH * NHEADS * NPAD * FHEAD * 2;// 12.98 MB
  const size_t SZ_BIAS = (size_t)NHEADS * NPAD * NPAD * 4;         // 53.5 MB
  __bf16* tokBf  = (__bf16*)(ws);
  __bf16* qkvWBf = (__bf16*)(ws + SZ_TOK);
  __bf16* prjWBf = (__bf16*)(ws + SZ_TOK + SZ_QKVW);
  __bf16* Qbf    = (__bf16*)(ws + SZ_TOK + SZ_QKVW + SZ_PRJW);
  __bf16* Kbf    = (__bf16*)(ws + SZ_TOK + SZ_QKVW + SZ_PRJW + SZ_QKV);
  __bf16* Vtbf   = (__bf16*)(ws + SZ_TOK + SZ_QKVW + SZ_PRJW + 2 * SZ_QKV);
  __bf16* ctxBf  = (__bf16*)(ws + SZ_TOK + SZ_QKVW + SZ_PRJW + 3 * SZ_QKV);
  float*  biasA  = (float*)(ws + 2 * SZ_TOK + SZ_QKVW + SZ_PRJW + 3 * SZ_QKV);
  (void)ws_size; (void)in_sizes; (void)n_in; (void)out_size;

  k_cvt_tokens<<<(BATCH * NPAD * FTOK) / 256, 256, 0, stream>>>(tokens, tokBf);
  k_cvt<<<(3 * FTOK * FTOK) / 256, 256, 0, stream>>>(qkv_w, qkvWBf, 3 * FTOK * FTOK);
  k_cvt<<<(FTOK * FTOK) / 256, 256, 0, stream>>>(proj_w, prjWBf, FTOK * FTOK);
  k_bias<<<(NHEADS * NPAD * NPAD) / 256, 256, 0, stream>>>(rpi, rtable, biasA);
  k_qkv<<<BATCH * NQT * 9, 128, 0, stream>>>(tokBf, qkvWBf, q_bias, v_bias,
                                             Qbf, Kbf, Vtbf);
  k_attn<<<BATCH * NHEADS * 17, 128, 0, stream>>>(Qbf, Kbf, Vtbf, biasA, ctxBf);
  k_proj<<<BATCH * NQT * 3, 128, 0, stream>>>(ctxBf, prjWBf, proj_b, out);
}
